// RNN_73151882986173
// MI455X (gfx1250) — matrix operand order repaired and checked by reading, not compiled, not run on hardware
//
#include <hip/hip_runtime.h>
#include <hip/hip_bf16.h>

// ---------------------------------------------------------------------------
// Problem constants (match reference)
// ---------------------------------------------------------------------------
#define T_STEPS   1024          // B*S
#define D_EMB     300
#define H1_DIM    1024
#define H2_DIM    512
#define IN1       (D_EMB + H1_DIM)   // 1324
#define IN2       (H1_DIM + H2_DIM)  // 1536
#define V_GLOBAL  50000
#define V_SENSE   25000
#define V_SENSE_PAD 25008            // pad to multiple of 16 for WMMA tiles

typedef __attribute__((ext_vector_type(8)))  float  v8f;
typedef __attribute__((ext_vector_type(16))) __bf16 v16bf;
typedef __attribute__((ext_vector_type(4)))  int    v4i;

#define GLOBAL_AS __attribute__((address_space(1)))
#define LDS_AS    __attribute__((address_space(3)))

// ---------------------------------------------------------------------------
// Kernel 1: fp32 -> bf16 weight conversion with zero row padding.
// ---------------------------------------------------------------------------
__global__ void cvt_f32_to_bf16_pad(const float* __restrict__ src,
                                    __hip_bfloat16* __restrict__ dst,
                                    size_t n_src, size_t n_dst) {
    size_t i = (size_t)blockIdx.x * blockDim.x + threadIdx.x;
    size_t stride = (size_t)gridDim.x * blockDim.x;
    for (; i < n_dst; i += stride) {
        float v = (i < n_src) ? src[i] : 0.0f;
        dst[i] = __float2bfloat16(v);
    }
}

// ---------------------------------------------------------------------------
// Kernel 2: sequential RNN scan. One workgroup of 1024 threads (32 waves on
// one WGP). State lives in LDS; W1/W2 rows streamed as b128 loads (hot in the
// 192MB L2 after step 0). Thread r owns row r of each layer. Emits h2 as bf16.
// ---------------------------------------------------------------------------
__global__ __launch_bounds__(1024)
void rnn_scan_kernel(const int* __restrict__ tokens,
                     const float* __restrict__ X,
                     const float* __restrict__ W1, const float* __restrict__ b1,
                     const float* __restrict__ W2, const float* __restrict__ b2,
                     __hip_bfloat16* __restrict__ h2b /* [T_STEPS][H2_DIM] */) {
    __shared__ __align__(16) float xin[IN1];   // [ e(300) | h1_prev(1024) ]
    __shared__ __align__(16) float x2 [IN2];   // [ h1_new(1024) | h2_prev(512) ]

    const int r = threadIdx.x;

    if (r < H1_DIM) xin[D_EMB + r] = 0.0f;
    if (r < H2_DIM) x2[H1_DIM + r] = 0.0f;
    __syncthreads();

    const float4* w1row = (const float4*)(W1 + (size_t)r * IN1);
    const float4* w2row = (r < H2_DIM) ? (const float4*)(W2 + (size_t)r * IN2) : nullptr;
    const float bb1 = b1[r];
    const float bb2 = (r < H2_DIM) ? b2[r] : 0.0f;

    for (int t = 0; t < T_STEPS; ++t) {
        const int tok = tokens[t];
        if (r < D_EMB) xin[r] = X[(size_t)tok * D_EMB + r];
        __syncthreads();

        // layer 1: h1n[r] = leaky( W1[r,:] . xin + b1[r] )
        float acc = bb1;
        const float4* xv = (const float4*)xin;
        #pragma unroll 4
        for (int j = 0; j < IN1 / 4; ++j) {
            float4 w = w1row[j];
            float4 x = xv[j];                 // LDS broadcast (uniform addr)
            acc += w.x * x.x + w.y * x.y + w.z * x.z + w.w * x.w;
        }
        float h1n = acc > 0.0f ? acc : 0.01f * acc;
        __syncthreads();                       // all reads of old h1 done
        xin[D_EMB + r] = h1n;
        x2[r] = h1n;
        __syncthreads();

        // layer 2: h2n[r] = leaky( W2[r,:] . x2 + b2[r] ), r < 512
        float h2n = 0.0f;
        if (r < H2_DIM) {
            float acc2 = bb2;
            const float4* x2v = (const float4*)x2;
            #pragma unroll 4
            for (int j = 0; j < IN2 / 4; ++j) {
                float4 w = w2row[j];
                float4 x = x2v[j];
                acc2 += w.x * x.x + w.y * x.y + w.z * x.z + w.w * x.w;
            }
            h2n = acc2 > 0.0f ? acc2 : 0.01f * acc2;
        }
        __syncthreads();                       // all reads of old h2 done
        if (r < H2_DIM) {
            x2[H1_DIM + r] = h2n;
            h2b[(size_t)t * H2_DIM + r] = __float2bfloat16(h2n);
        }
        __syncthreads();
    }
}

// ---------------------------------------------------------------------------
// Kernel 3: head GEMM, bf16 WMMA with fp32 accumulate.
//   out[t, v] = sum_k h2b[t,k] * Wb[v,k] + bias[v]
//
// Block = 256 threads = 8 waves, all sharing rows [m0, m0+32). The shared
// 32x512 bf16 A tile (32 KB) is staged into LDS ONCE per block with
// GLOBAL_LOAD_ASYNC_TO_LDS_B128 (ASYNCcnt; fenced by s_wait_asynccnt + one
// barrier). Each wave then owns a 32(M) x 16(N) output tile: two stacked
// 16x16 accumulators sharing one B fragment, K-loop of 16
// v_wmma_f32_16x16x32_bf16. A fragments come from LDS (ds_load_b128), B
// streams from L2 (global_load_b128 + global_prefetch_b8).
//
// Fragment layouts per CDNA5 ISA §7.12.2 (wave32):
//   A 16x32 bf16 : lane (l&15)=M; elems 0..7 = K kb..kb+7, 8..15 = kb+16..+23,
//                  kb = (l>>4)*8  -> two 16B loads per fragment.
//   B 32x16 bf16 : lane (l&15)=N; elems 0..7 = K kb..kb+7, 8..15 = kb+16..+23 (as A)
//                  -> two 16B loads of the (row-major) weight row.
//   C 16x16 f32  : elem i -> M = i + (l>>4)*8, N = (l&15).
// ---------------------------------------------------------------------------
union Frag16 { uint4 q[2]; v16bf v; };
union Acc8   { v8f v; float f[8]; };

__global__ __launch_bounds__(256)
void head_gemm_kernel(const __hip_bfloat16* __restrict__ h2b,
                      const __hip_bfloat16* __restrict__ Wb,   // padded rows
                      const float* __restrict__ bias,
                      float* __restrict__ out,
                      int V,          // true vocab (bounds for stores)
                      int ntiles) {   // ceil(V/16)
    __shared__ __align__(16) __hip_bfloat16 As[32 * H2_DIM];   // 32 KB A tile

    const int tid  = threadIdx.x;
    const int wave = tid >> 5;
    const int lane = tid & 31;
    const int m0   = blockIdx.y * 32;               // rows m0..m0+31 (shared)
    const int nt   = blockIdx.x * 8 + wave;

    // ---- async stage of the block-shared A tile: 2048 x 16B chunks ----
    {
        const __hip_bfloat16* gsrc = h2b + (size_t)m0 * H2_DIM;
        #pragma unroll
        for (int p = 0; p < 8; ++p) {
            const int c = tid + p * 256;            // 16-byte chunk id
            __builtin_amdgcn_global_load_async_to_lds_b128(
                (GLOBAL_AS v4i*)(gsrc + (size_t)c * 8),
                (LDS_AS v4i*)(As + (size_t)c * 8),
                0, 0);
        }
        asm volatile("s_wait_asynccnt 0x0" ::: "memory");
    }
    __syncthreads();                                // only barrier in kernel

    if (nt >= ntiles) return;                       // after barrier: safe

    const int m1  = m0 + 16;
    const int v0  = nt * 16;
    const int lm  = lane & 15;
    const int lhi = lane >> 4;
    const int kbA = lhi * 8;                        // A lane K base
    const int kbB = lhi * 16;                       // B prefetch lane K offset

    const __hip_bfloat16* arow0 = As + (size_t)lm * H2_DIM;         // LDS
    const __hip_bfloat16* arow1 = As + (size_t)(lm + 16) * H2_DIM;  // LDS
    const __hip_bfloat16* brow  = Wb + (size_t)(v0 + lm) * H2_DIM;  // global

    v8f c0 = {};
    v8f c1 = {};

    #pragma unroll 4
    for (int k0 = 0; k0 < H2_DIM; k0 += 32) {
        Frag16 a0, a1, b;
        // A fragments from LDS (ds_load_b128)
        a0.q[0] = *(const uint4*)(arow0 + k0 + kbA);
        a0.q[1] = *(const uint4*)(arow0 + k0 + kbA + 16);
        a1.q[0] = *(const uint4*)(arow1 + k0 + kbA);
        a1.q[1] = *(const uint4*)(arow1 + k0 + kbA + 16);
        // B fragment: same K layout as A (kbA..+7, kbA+16..+23) of the weight row (global)
        b.q[0]  = *(const uint4*)(brow + k0 + kbA);
        b.q[1]  = *(const uint4*)(brow + k0 + kbA + 16);
        if (k0 + 32 < H2_DIM)
            __builtin_prefetch(brow + k0 + 32 + kbB, 0, 1);

        c0 = __builtin_amdgcn_wmma_f32_16x16x32_bf16(
                 false, a0.v, false, b.v, (short)0, c0, false, false);
        c1 = __builtin_amdgcn_wmma_f32_16x16x32_bf16(
                 false, a1.v, false, b.v, (short)0, c1, false, false);
    }

    // Epilogue: +bias, bounds-checked store (lane column uniform over elems).
    const int col = v0 + lm;
    if (col < V) {
        const float bv = bias[col];
        Acc8 r0, r1; r0.v = c0; r1.v = c1;
        const int rb = lhi * 8;
        #pragma unroll
        for (int i = 0; i < 8; ++i) {
            out[(size_t)(m0 + rb + i) * V + col] = r0.f[i] + bv;
            out[(size_t)(m1 + rb + i) * V + col] = r1.f[i] + bv;
        }
    }
}

// ---------------------------------------------------------------------------
// Kernel 4: row-wise log-softmax, one 256-thread block per row.
// ---------------------------------------------------------------------------
__global__ __launch_bounds__(256)
void log_softmax_kernel(float* __restrict__ out, int V) {
    __shared__ float sm[256];
    __shared__ float ss[256];
    float* row = out + (size_t)blockIdx.x * V;
    const int tid = threadIdx.x;

    float m = -3.4e38f, s = 0.0f;
    for (int j = tid; j < V; j += 256) {
        float x = row[j];
        float nm = fmaxf(m, x);
        s = s * __expf(m - nm) + __expf(x - nm);
        m = nm;
    }
    sm[tid] = m; ss[tid] = s;
    __syncthreads();
    for (int off = 128; off > 0; off >>= 1) {
        if (tid < off) {
            float m2 = sm[tid + off], s2 = ss[tid + off];
            float nm = fmaxf(sm[tid], m2);
            ss[tid] = ss[tid] * __expf(sm[tid] - nm) + s2 * __expf(m2 - nm);
            sm[tid] = nm;
        }
        __syncthreads();
    }
    const float lse = sm[0] + __logf(ss[0]);
    for (int j = tid; j < V; j += 256)
        row[j] = row[j] - lse;
}

// ---------------------------------------------------------------------------
// Launcher
// ---------------------------------------------------------------------------
extern "C" void kernel_launch(void* const* d_in, const int* in_sizes, int n_in,
                              void* d_out, int out_size, void* d_ws, size_t ws_size,
                              hipStream_t stream) {
    const int*   tokens = (const int*)  d_in[0];
    const float* X      = (const float*)d_in[1];
    const float* W1     = (const float*)d_in[2];
    const float* b1     = (const float*)d_in[3];
    const float* W2     = (const float*)d_in[4];
    const float* b2     = (const float*)d_in[5];
    const float* Wg     = (const float*)d_in[6];
    const float* bg     = (const float*)d_in[7];
    const float* Ws     = (const float*)d_in[8];
    const float* bs     = (const float*)d_in[9];

    // workspace carve (bytes), 256B-aligned regions
    char* ws = (char*)d_ws;
    const size_t wg_elems  = (size_t)V_GLOBAL    * H2_DIM;
    const size_t wsp_elems = (size_t)V_SENSE_PAD * H2_DIM;
    const size_t ws_elems  = (size_t)V_SENSE     * H2_DIM;
    __hip_bfloat16* Wg_b = (__hip_bfloat16*)ws;
    size_t off = (wg_elems * 2 + 255) & ~(size_t)255;
    __hip_bfloat16* Ws_b = (__hip_bfloat16*)(ws + off);
    off += (wsp_elems * 2 + 255) & ~(size_t)255;
    __hip_bfloat16* h2b  = (__hip_bfloat16*)(ws + off);

    // 1) weight conversion to bf16 (grid-stride)
    {
        int blocks = 4096;
        cvt_f32_to_bf16_pad<<<blocks, 256, 0, stream>>>(Wg, Wg_b, wg_elems, wg_elems);
        cvt_f32_to_bf16_pad<<<blocks, 256, 0, stream>>>(Ws, Ws_b, ws_elems, wsp_elems);
    }

    // 2) sequential RNN scan (single workgroup — latency-bound recurrence)
    rnn_scan_kernel<<<1, 1024, 0, stream>>>(tokens, X, W1, b1, W2, b2, h2b);

    // 3) WMMA head GEMMs (+bias)
    float* outG = (float*)d_out;
    float* outS = outG + (size_t)T_STEPS * V_GLOBAL;
    {
        const int ntg = V_GLOBAL / 16;                   // 3125
        dim3 grid((ntg + 7) / 8, T_STEPS / 32);
        head_gemm_kernel<<<grid, 256, 0, stream>>>(h2b, Wg_b, bg, outG, V_GLOBAL, ntg);
    }
    {
        const int nts = (V_SENSE + 15) / 16;             // 1563 (padded rows cover it)
        dim3 grid((nts + 7) / 8, T_STEPS / 32);
        head_gemm_kernel<<<grid, 256, 0, stream>>>(h2b, Ws_b, bs, outS, V_SENSE, nts);
    }

    // 4) log-softmax per row, per head
    log_softmax_kernel<<<T_STEPS, 256, 0, stream>>>(outG, V_GLOBAL);
    log_softmax_kernel<<<T_STEPS, 256, 0, stream>>>(outS, V_SENSE);
}
